// Decoder_28329604284728
// MI455X (gfx1250) — matrix operand order repaired and checked by reading, not compiled, not run on hardware
//
#include <hip/hip_runtime.h>
#include <math.h>

typedef __attribute__((ext_vector_type(16))) _Float16 v16h;
typedef __attribute__((ext_vector_type(8)))  _Float16 v8h;
typedef __attribute__((ext_vector_type(8)))  float    v8f;

#define NUM_MELS 80
#define ENC_DIM  512
#define ATT_RNN  1024
#define DEC_RNN  1024
#define PRENET_D 256
#define ATT_DIM  128
#define LOC_F    32
#define LOC_K    31
#define BB       32
#define T_IN     192
#define T_MEL    400
#define GATES_N  4096
#define KC_ATT   (PRENET_D + ENC_DIM + ATT_RNN)   // 1792
#define KC_DEC   (ATT_RNN + ENC_DIM + DEC_RNN)    // 2560
#define CHUNK_K     256
#define CHUNK_TILES (CHUNK_K / 32)                // 8 k-tiles per LDS chunk

__device__ __forceinline__ float sigf(float x) { return 1.0f / (1.0f + __expf(-x)); }

// ---------------------------------------------------------------------------
// Zero-fill
__global__ void zero_f32(float* p, int n) {
    int i = blockIdx.x * blockDim.x + threadIdx.x;
    if (i < n) p[i] = 0.0f;
}

// ---------------------------------------------------------------------------
// Pack combined LSTM weight [wih | whh] (row-major, rows = 4096 gate outputs)
// into f16 B-fragment tiles for v_wmma_f32_16x16x32_f16.
// Tile (kt, nt): 512 halves; element at lane*16+e maps to
//   k = kt*32 + (lane>=16 ? 16 : 0) + e,   n = nt*16 + (lane & 15)
__global__ void pack_lstm_weights(const float* __restrict__ wih,
                                  const float* __restrict__ whh,
                                  int Kx, int Kh, int Kc,
                                  _Float16* __restrict__ Wp) {
    size_t idx = (size_t)blockIdx.x * blockDim.x + threadIdx.x;
    size_t total = (size_t)Kc * GATES_N;
    if (idx >= total) return;
    int e    = (int)(idx & 15);
    int lane = (int)((idx >> 4) & 31);
    size_t tile = idx >> 9;
    int nt = (int)(tile & 255);
    int kt = (int)(tile >> 8);
    int k = kt * 32 + ((lane >> 4) ? 16 : 0) + e;
    int n = nt * 16 + (lane & 15);
    float w = (k < Kx) ? wih[(size_t)n * Kx + k] : whh[(size_t)n * Kh + (k - Kx)];
    Wp[idx] = (_Float16)w;
}

__global__ void combine_bias(const float* a, const float* b, float* o, int n) {
    int i = blockIdx.x * blockDim.x + threadIdx.x;
    if (i < n) o[i] = a[i] + b[i];
}

// ---------------------------------------------------------------------------
// Prenet (all 400 steps up front). Step t input: zeros if t==0 else frame t-1.
__global__ void prenet1(const float* __restrict__ din,   // (B,80,T_MEL)
                        const float* __restrict__ w1,    // (256,80)
                        float* __restrict__ h1) {        // (T_MEL,B,256)
    int idx = blockIdx.x * blockDim.x + threadIdx.x;
    if (idx >= T_MEL * BB * PRENET_D) return;
    int p = idx % PRENET_D;
    int b = (idx / PRENET_D) % BB;
    int t = idx / (PRENET_D * BB);
    float s = 0.0f;
    if (t > 0) {
        const float* fr = din + (size_t)b * NUM_MELS * T_MEL + (t - 1);
        const float* wr = w1 + (size_t)p * NUM_MELS;
        for (int m = 0; m < NUM_MELS; ++m) s += wr[m] * fr[(size_t)m * T_MEL];
    }
    h1[idx] = fmaxf(s, 0.0f);
}

__global__ void prenet2(const float* __restrict__ h1,
                        const float* __restrict__ w2,    // (256,256)
                        float* __restrict__ px) {        // (T_MEL,B,256)
    int idx = blockIdx.x * blockDim.x + threadIdx.x;
    if (idx >= T_MEL * BB * PRENET_D) return;
    int p  = idx % PRENET_D;
    int tb = idx / PRENET_D;
    const float* hr = h1 + (size_t)tb * PRENET_D;
    const float* wr = w2 + (size_t)p * PRENET_D;
    float s = 0.0f;
    for (int q = 0; q < PRENET_D; ++q) s += wr[q] * hr[q];
    px[idx] = fmaxf(s, 0.0f);
}

// ---------------------------------------------------------------------------
// pmem[b,t,a] = sum_d memory[b,t,d] * memory_w[a,d]
__global__ void pmem_kernel(const float* __restrict__ memory,
                            const float* __restrict__ memw,
                            float* __restrict__ pmem) {
    int idx = blockIdx.x * blockDim.x + threadIdx.x;
    if (idx >= BB * T_IN * ATT_DIM) return;
    int a  = idx % ATT_DIM;
    int bt = idx / ATT_DIM;
    const float* mr = memory + (size_t)bt * ENC_DIM;
    const float* wr = memw + (size_t)a * ENC_DIM;
    float s = 0.0f;
    for (int d = 0; d < ENC_DIM; ++d) s += mr[d] * wr[d];
    pmem[idx] = s;
}

// ---------------------------------------------------------------------------
// Pack z = [a(na) | b(nb) | c(nc)] per row (32 rows) into f16, row-major.
__global__ void pack_z3(const float* __restrict__ pa, int na,
                        const float* __restrict__ pb, int nb,
                        const float* __restrict__ pc, int nc,
                        _Float16* __restrict__ z) {
    int Kc = na + nb + nc;
    int idx = blockIdx.x * blockDim.x + threadIdx.x;
    if (idx >= BB * Kc) return;
    int row = idx / Kc;
    int k   = idx % Kc;
    float v;
    if (k < na)            v = pa[(size_t)row * na + k];
    else if (k < na + nb)  v = pb[(size_t)row * nb + (k - na)];
    else                   v = pc[(size_t)row * nc + (k - na - nb)];
    z[idx] = (_Float16)v;
}

// ---------------------------------------------------------------------------
// gates(32 x 4096) = z(32 x KC) @ Wp + bias, via v_wmma_f32_16x16x32_f16.
// KC is a compile-time template parameter (1792 / 2560) so every chunk/tile
// bound resolves statically: the 8-tile chunk body fully unrolls, B-tile
// prefetch is unconditional in the steady state, and the last chunk is peeled.
// Grid: 32 blocks x 8 waves = 256 waves; wave handles one 16-col N-tile and
// both 16-row M-tiles. z staged via double-buffered LDS (ds_load_b128 A-frags).
template <int KC>
__global__ void __launch_bounds__(256)
lstm_gates_wmma(const _Float16* __restrict__ z,   // 32 x KC row-major f16
                const _Float16* __restrict__ Wp,  // packed B-fragment tiles
                const float* __restrict__ bias,   // 4096 (bih+bhh)
                float* __restrict__ gates) {      // 32 x 4096
    constexpr int NCHUNKS = KC / CHUNK_K;         // 7 (att) or 10 (dec)

    __shared__ _Float16 zsh[2][BB * CHUNK_K];     // 2 x 16 KB

    int tid  = threadIdx.x;
    int lane = tid & 31;
    int wv   = tid >> 5;                          // 0..7
    int nt   = blockIdx.x * 8 + wv;               // 0..255
    int col  = lane & 15;
    int hi   = lane >> 4;

    // Cooperative staging: 256 threads x 32 halves = one 32x256 chunk.
    int srow = tid >> 3;                          // 0..31
    int sseg = (tid & 7) * 32;                    // 0,32,...,224
    const _Float16* zsrc = z + (size_t)srow * KC + sseg;
    {
        _Float16* dst = &zsh[0][srow * CHUNK_K + sseg];
#pragma unroll
        for (int u = 0; u < 32; u += 8)
            *(v8h*)(dst + u) = *(const v8h*)(zsrc + u);
    }
    __syncthreads();

    const _Float16* wpbase = Wp + (size_t)nt * 512 + lane * 16;

    v8f acc0 = {}, acc1 = {};
    union AFrag { v16h v; v8h h[2]; };

    v16h bfrag = *(const v16h*)(wpbase);          // B tile for kt = 0

    // Steady-state chunks: stage chunk c+1, compute chunk c; next B tile
    // always exists.
    for (int c = 0; c < NCHUNKS - 1; ++c) {
        {
            const _Float16* src = zsrc + (c + 1) * CHUNK_K;
            _Float16* dst = &zsh[(c + 1) & 1][srow * CHUNK_K + sseg];
#pragma unroll
            for (int u = 0; u < 32; u += 8)
                *(v8h*)(dst + u) = *(const v8h*)(src + u);
        }
        const _Float16* zs = &zsh[c & 1][0];
        const _Float16* z0 = zs + col * CHUNK_K + hi * 16;         // rows 0-15
        const _Float16* z1 = zs + (col + 16) * CHUNK_K + hi * 16;  // rows 16-31
        const _Float16* wpc = wpbase + (size_t)c * CHUNK_TILES * 256 * 512;
#pragma unroll
        for (int ktl = 0; ktl < CHUNK_TILES; ++ktl) {
            v16h bnext = *(const v16h*)(wpc + (size_t)(ktl + 1) * 256 * 512);
            AFrag a0, a1;
            a0.h[0] = *(const v8h*)(z0 + ktl * 32);
            a0.h[1] = *(const v8h*)(z0 + ktl * 32 + 8);
            a1.h[0] = *(const v8h*)(z1 + ktl * 32);
            a1.h[1] = *(const v8h*)(z1 + ktl * 32 + 8);
            acc0 = __builtin_amdgcn_wmma_f32_16x16x32_f16(false, a0.v, false, bfrag,
                                                          (short)0, acc0, false, false);
            acc1 = __builtin_amdgcn_wmma_f32_16x16x32_f16(false, a1.v, false, bfrag,
                                                          (short)0, acc1, false, false);
            bfrag = bnext;
        }
        __syncthreads();   // staging of next chunk complete before reuse
    }

    // Final chunk (peeled): no staging; last tile has no successor B tile.
    {
        constexpr int c = NCHUNKS - 1;
        const _Float16* zs = &zsh[c & 1][0];
        const _Float16* z0 = zs + col * CHUNK_K + hi * 16;
        const _Float16* z1 = zs + (col + 16) * CHUNK_K + hi * 16;
        const _Float16* wpc = wpbase + (size_t)c * CHUNK_TILES * 256 * 512;
#pragma unroll
        for (int ktl = 0; ktl < CHUNK_TILES; ++ktl) {
            v16h bnext = bfrag;
            if (ktl + 1 < CHUNK_TILES)   // compile-time after unroll
                bnext = *(const v16h*)(wpc + (size_t)(ktl + 1) * 256 * 512);
            AFrag a0, a1;
            a0.h[0] = *(const v8h*)(z0 + ktl * 32);
            a0.h[1] = *(const v8h*)(z0 + ktl * 32 + 8);
            a1.h[0] = *(const v8h*)(z1 + ktl * 32);
            a1.h[1] = *(const v8h*)(z1 + ktl * 32 + 8);
            acc0 = __builtin_amdgcn_wmma_f32_16x16x32_f16(false, a0.v, false, bfrag,
                                                          (short)0, acc0, false, false);
            acc1 = __builtin_amdgcn_wmma_f32_16x16x32_f16(false, a1.v, false, bfrag,
                                                          (short)0, acc1, false, false);
            bfrag = bnext;
        }
    }

    int n = nt * 16 + col;
    float bv = bias[n];
#pragma unroll
    for (int r = 0; r < 8; ++r) {
        int m = hi * 8 + r;
        gates[(size_t)m * GATES_N + n]        = acc0[r] + bv;
        gates[(size_t)(m + 16) * GATES_N + n] = acc1[r] + bv;
    }
}

// ---------------------------------------------------------------------------
// LSTM pointwise: i,f,g,o blocks of 1024 each. Updates h, c in place.
__global__ void lstm_pointwise(const float* __restrict__ gates,
                               float* __restrict__ h, float* __restrict__ c) {
    int idx = blockIdx.x * blockDim.x + threadIdx.x;
    if (idx >= BB * 1024) return;
    int b = idx / 1024;
    int j = idx % 1024;
    const float* g = gates + (size_t)b * GATES_N;
    float ig = sigf(g[j]);
    float fg = sigf(g[1024 + j]);
    float gg = tanhf(g[2048 + j]);
    float og = sigf(g[3072 + j]);
    float c2 = fg * c[idx] + ig * gg;
    c[idx] = c2;
    h[idx] = og * tanhf(c2);
}

// ---------------------------------------------------------------------------
// pq[b,a] = sum_j ah[b,j] * query_w[a,j]
__global__ void attn_pq(const float* __restrict__ ah,
                        const float* __restrict__ qw,
                        float* __restrict__ pq) {
    int idx = blockIdx.x * blockDim.x + threadIdx.x;
    if (idx >= BB * ATT_DIM) return;
    int a = idx % ATT_DIM;
    int b = idx / ATT_DIM;
    const float* hr = ah + (size_t)b * ATT_RNN;
    const float* wr = qw + (size_t)a * ATT_RNN;
    float s = 0.0f;
    for (int j = 0; j < ATT_RNN; ++j) s += hr[j] * wr[j];
    pq[idx] = s;
}

// ---------------------------------------------------------------------------
// Energies: location conv (channels: aw, awc) -> loc_dense -> tanh -> v dot.
__global__ void attn_energy(const float* __restrict__ pq,      // 32x128
                            const float* __restrict__ pmem,    // 32x192x128
                            const float* __restrict__ aw,      // 32x192
                            const float* __restrict__ awc,     // 32x192
                            const float* __restrict__ convw,   // 32x2x31
                            const float* __restrict__ densew,  // 128x32
                            const float* __restrict__ vw,      // 128
                            const int*   __restrict__ mlen,    // 32
                            float* __restrict__ e) {           // 32x192
    int idx = blockIdx.x * blockDim.x + threadIdx.x;
    if (idx >= BB * T_IN) return;
    int t = idx % T_IN;
    int b = idx / T_IN;
    if (t >= mlen[b]) { e[idx] = -3.0e38f; return; }
    const int pad = (LOC_K - 1) / 2;
    float conv[LOC_F];
    for (int f = 0; f < LOC_F; ++f) {
        float s = 0.0f;
        for (int c = 0; c < 2; ++c) {
            const float* src = (c == 0) ? aw : awc;
            const float* wr = convw + ((size_t)f * 2 + c) * LOC_K;
            for (int k = 0; k < LOC_K; ++k) {
                int tt = t + k - pad;
                if (tt >= 0 && tt < T_IN) s += src[(size_t)b * T_IN + tt] * wr[k];
            }
        }
        conv[f] = s;
    }
    const float* pqr = pq + (size_t)b * ATT_DIM;
    const float* pmr = pmem + (size_t)idx * ATT_DIM;
    float en = 0.0f;
    for (int a = 0; a < ATT_DIM; ++a) {
        const float* dr = densew + (size_t)a * LOC_F;
        float pa = 0.0f;
        for (int f = 0; f < LOC_F; ++f) pa += dr[f] * conv[f];
        en += vw[a] * tanhf(pqr[a] + pa + pmr[a]);
    }
    e[idx] = en;
}

// ---------------------------------------------------------------------------
// Per-batch softmax over T_IN, update aw/awc, context matvec, emit alignment.
__global__ void __launch_bounds__(256)
softmax_ctx(const float* __restrict__ e,        // 32x192
            const float* __restrict__ memory,   // 32x192x512
            float* __restrict__ aw,
            float* __restrict__ awc,
            float* __restrict__ ctx,            // 32x512
            float* __restrict__ alignOut,       // base of alignments
            int step) {
    __shared__ float sdata[256];
    __shared__ float saw[T_IN];
    int b = blockIdx.x;
    int tid = threadIdx.x;
    float v = (tid < T_IN) ? e[(size_t)b * T_IN + tid] : -3.0e38f;
    sdata[tid] = v;
    __syncthreads();
    for (int s = 128; s > 0; s >>= 1) {
        if (tid < s) sdata[tid] = fmaxf(sdata[tid], sdata[tid + s]);
        __syncthreads();
    }
    float mx = sdata[0];
    __syncthreads();
    float p = (tid < T_IN) ? __expf(v - mx) : 0.0f;
    sdata[tid] = p;
    __syncthreads();
    for (int s = 128; s > 0; s >>= 1) {
        if (tid < s) sdata[tid] += sdata[tid + s];
        __syncthreads();
    }
    float inv = 1.0f / sdata[0];
    if (tid < T_IN) {
        float w = p * inv;
        saw[tid] = w;
        aw[(size_t)b * T_IN + tid] = w;
        awc[(size_t)b * T_IN + tid] += w;
        alignOut[((size_t)b * T_MEL + step) * T_IN + tid] = w;
    }
    __syncthreads();
    for (int d = tid; d < ENC_DIM; d += 256) {
        float s = 0.0f;
        const float* mr = memory + (size_t)b * T_IN * ENC_DIM + d;
        for (int t = 0; t < T_IN; ++t) s += saw[t] * mr[(size_t)t * ENC_DIM];
        ctx[(size_t)b * ENC_DIM + d] = s;
    }
}

// ---------------------------------------------------------------------------
// mel = [dh|ctx] @ proj_w.T + proj_b ; gate = [dh|ctx] @ gate_w.T + gate_b
__global__ void projection(const float* __restrict__ dh,
                           const float* __restrict__ ctx,
                           const float* __restrict__ projw,  // 80x1536
                           const float* __restrict__ projb,  // 80
                           const float* __restrict__ gatew,  // 1536
                           const float* __restrict__ gateb,  // 1
                           float* __restrict__ melOut,       // (B,80,T_MEL)
                           float* __restrict__ gateOut,      // (B,T_MEL)
                           int step) {
    int idx = blockIdx.x * blockDim.x + threadIdx.x;
    if (idx >= BB * (NUM_MELS + 1)) return;
    int o = idx % (NUM_MELS + 1);
    int b = idx / (NUM_MELS + 1);
    const float* hr = dh + (size_t)b * DEC_RNN;
    const float* cr = ctx + (size_t)b * ENC_DIM;
    const float* wr = (o < NUM_MELS) ? (projw + (size_t)o * (DEC_RNN + ENC_DIM)) : gatew;
    float s = 0.0f;
    for (int j = 0; j < DEC_RNN; ++j) s += wr[j] * hr[j];
    for (int j = 0; j < ENC_DIM; ++j) s += wr[DEC_RNN + j] * cr[j];
    if (o < NUM_MELS)
        melOut[((size_t)b * NUM_MELS + o) * T_MEL + step] = s + projb[o];
    else
        gateOut[(size_t)b * T_MEL + step] = s + gateb[0];
}

// ---------------------------------------------------------------------------
static inline int cdiv(long long a, int b) { return (int)((a + b - 1) / b); }

extern "C" void kernel_launch(void* const* d_in, const int* in_sizes, int n_in,
                              void* d_out, int out_size, void* d_ws, size_t ws_size,
                              hipStream_t stream) {
    const float* memory  = (const float*)d_in[0];
    const float* dec_in  = (const float*)d_in[1];
    const int*   mlen    = (const int*)d_in[3];
    const float* w1      = (const float*)d_in[4];
    const float* w2      = (const float*)d_in[5];
    const float* att_wih = (const float*)d_in[6];
    const float* att_whh = (const float*)d_in[7];
    const float* att_bih = (const float*)d_in[8];
    const float* att_bhh = (const float*)d_in[9];
    const float* query_w = (const float*)d_in[10];
    const float* memw    = (const float*)d_in[11];
    const float* vw      = (const float*)d_in[12];
    const float* convw   = (const float*)d_in[13];
    const float* densew  = (const float*)d_in[14];
    const float* dec_wih = (const float*)d_in[15];
    const float* dec_whh = (const float*)d_in[16];
    const float* dec_bih = (const float*)d_in[17];
    const float* dec_bhh = (const float*)d_in[18];
    const float* projw   = (const float*)d_in[19];
    const float* projb   = (const float*)d_in[20];
    const float* gatew   = (const float*)d_in[21];
    const float* gateb   = (const float*)d_in[22];

    float* out = (float*)d_out;
    float* melOut   = out;                              // B*80*T_MEL
    float* gateOut  = out + (size_t)BB * NUM_MELS * T_MEL;
    float* alignOut = gateOut + (size_t)BB * T_MEL;

    // Workspace carve-up (256B aligned)
    char* ws = (char*)d_ws;
    size_t off = 0;
    auto alloc = [&](size_t bytes) {
        off = (off + 255) & ~(size_t)255;
        size_t o = off;
        off += bytes;
        return o;
    };
    _Float16* WpAtt = (_Float16*)(ws + alloc((size_t)KC_ATT * GATES_N * 2));
    _Float16* WpDec = (_Float16*)(ws + alloc((size_t)KC_DEC * GATES_N * 2));
    float* h1   = (float*)(ws + alloc((size_t)T_MEL * BB * PRENET_D * 4));
    float* px   = (float*)(ws + alloc((size_t)T_MEL * BB * PRENET_D * 4));
    float* pmem = (float*)(ws + alloc((size_t)BB * T_IN * ATT_DIM * 4));
    float* biasA = (float*)(ws + alloc(GATES_N * 4));
    float* biasD = (float*)(ws + alloc(GATES_N * 4));
    float* gatesBuf = (float*)(ws + alloc((size_t)BB * GATES_N * 4));
    _Float16* zbuf = (_Float16*)(ws + alloc((size_t)BB * KC_DEC * 2));
    float* pq   = (float*)(ws + alloc((size_t)BB * ATT_DIM * 4));
    float* eBuf = (float*)(ws + alloc((size_t)BB * T_IN * 4));
    // Contiguous recurrent state block (zeroed each launch)
    const int stateFloats = BB * (4 * 1024 + 2 * T_IN + ENC_DIM);
    float* state = (float*)(ws + alloc((size_t)stateFloats * 4));
    float* ah  = state;
    float* ac  = ah + BB * 1024;
    float* dh  = ac + BB * 1024;
    float* dc  = dh + BB * 1024;
    float* aw  = dc + BB * 1024;
    float* awc = aw + BB * T_IN;
    float* ctx = awc + BB * T_IN;

    // ---- Preprocessing (deterministic, every call) ----
    zero_f32<<<cdiv(stateFloats, 256), 256, 0, stream>>>(state, stateFloats);

    pack_lstm_weights<<<cdiv((long long)KC_ATT * GATES_N, 256), 256, 0, stream>>>(
        att_wih, att_whh, PRENET_D + ENC_DIM, ATT_RNN, KC_ATT, WpAtt);
    pack_lstm_weights<<<cdiv((long long)KC_DEC * GATES_N, 256), 256, 0, stream>>>(
        dec_wih, dec_whh, ATT_RNN + ENC_DIM, DEC_RNN, KC_DEC, WpDec);
    combine_bias<<<cdiv(GATES_N, 256), 256, 0, stream>>>(att_bih, att_bhh, biasA, GATES_N);
    combine_bias<<<cdiv(GATES_N, 256), 256, 0, stream>>>(dec_bih, dec_bhh, biasD, GATES_N);

    prenet1<<<cdiv((long long)T_MEL * BB * PRENET_D, 256), 256, 0, stream>>>(dec_in, w1, h1);
    prenet2<<<cdiv((long long)T_MEL * BB * PRENET_D, 256), 256, 0, stream>>>(h1, w2, px);
    pmem_kernel<<<cdiv((long long)BB * T_IN * ATT_DIM, 256), 256, 0, stream>>>(memory, memw, pmem);

    // ---- Sequential decode loop ----
    for (int t = 0; t < T_MEL; ++t) {
        // Attention LSTM: z = [prenet_x(t) | ctx | ah]
        pack_z3<<<cdiv((long long)BB * KC_ATT, 256), 256, 0, stream>>>(
            px + (size_t)t * BB * PRENET_D, PRENET_D, ctx, ENC_DIM, ah, ATT_RNN, zbuf);
        lstm_gates_wmma<KC_ATT><<<32, 256, 0, stream>>>(zbuf, WpAtt, biasA, gatesBuf);
        lstm_pointwise<<<cdiv(BB * 1024, 256), 256, 0, stream>>>(gatesBuf, ah, ac);

        // Attention
        attn_pq<<<cdiv(BB * ATT_DIM, 256), 256, 0, stream>>>(ah, query_w, pq);
        attn_energy<<<cdiv(BB * T_IN, 128), 128, 0, stream>>>(
            pq, pmem, aw, awc, convw, densew, vw, mlen, eBuf);
        softmax_ctx<<<BB, 256, 0, stream>>>(eBuf, memory, aw, awc, ctx, alignOut, t);

        // Decoder LSTM: z = [ah | ctx | dh]
        pack_z3<<<cdiv((long long)BB * KC_DEC, 256), 256, 0, stream>>>(
            ah, ATT_RNN, ctx, ENC_DIM, dh, DEC_RNN, zbuf);
        lstm_gates_wmma<KC_DEC><<<32, 256, 0, stream>>>(zbuf, WpDec, biasD, gatesBuf);
        lstm_pointwise<<<cdiv(BB * 1024, 256), 256, 0, stream>>>(gatesBuf, dh, dc);

        // Outputs
        projection<<<cdiv(BB * (NUM_MELS + 1), 128), 128, 0, stream>>>(
            dh, ctx, projw, projb, gatew, gateb, melOut, gateOut, t);
    }
}
